// PointPillarMash_41661182771846
// MI455X (gfx1250) — compile-verified, matrix operand order checked by reading, not run on hardware
//
#include <hip/hip_runtime.h>
#include <hip/hip_bf16.h>
#include <math.h>

// ---------------- problem constants ----------------
#define HH    48
#define WW    176
#define HW    8448          // HH*WW
#define KCH   8449          // HW+1 correlation channels
#define CDIM  256
#define KD    64
#define ANC   2
#define NREG  14            // 7*ANC

typedef __attribute__((ext_vector_type(16))) _Float16 v16h;
typedef __attribute__((ext_vector_type(8)))  float    v8f;

union Frag16 { v16h v; uint4 q[2]; };
union FragC  { v8f  v; float f[8]; };
union Pack8h { uint4 u; _Float16 h[8]; };

#define LDS_STRIDE 72   // 64 halves + 16B pad, keeps 16B alignment

// ---- CDNA5 async global->LDS (ASYNCcnt-tracked DMA-style copy) ----
#if defined(__gfx1250__) && __has_builtin(__builtin_amdgcn_global_load_async_to_lds_b128)
#define USE_ASYNC_LDS 1
typedef int v4i __attribute__((ext_vector_type(4)));
typedef __attribute__((address_space(1))) v4i gv4i;   // global int4 vector
typedef __attribute__((address_space(3))) v4i lv4i;   // LDS int4 vector
__device__ __forceinline__ void async_copy16(const void* g, void* l) {
    __builtin_amdgcn_global_load_async_to_lds_b128((gv4i*)g, (lv4i*)l, 0, 0);
}
__device__ __forceinline__ void async_wait0() {
#if __has_builtin(__builtin_amdgcn_s_wait_asynccnt)
    __builtin_amdgcn_s_wait_asynccnt(0);
#else
    asm volatile("s_wait_asynccnt 0" ::: "memory");
#endif
}
#else
#define USE_ASYNC_LDS 0
#endif

// Load one 16x32 f16 WMMA operand fragment from an LDS tile stored
// row-major (row = M or N index, LDS_STRIDE halves per row).
// Layout per CDNA5 ISA 7.12.2 (16-bit A 16x32 / B 32x16 mirrored):
//   lanes 0-15 : row = lane,     halves = K[ks..ks+7]   ++ K[ks+16..ks+23]
//   lanes 16-31: row = lane-16,  halves = K[ks+8..ks+15]++ K[ks+24..ks+31]
__device__ inline v16h lds_frag(const _Float16* base, int rowBase, int ks, int lane) {
    int row = rowBase + (lane & 15);
    int hi  = (lane >> 4) << 3;
    const _Float16* p = base + row * LDS_STRIDE + ks + hi;
    Frag16 f;
    f.q[0] = *(const uint4*)(p);
    f.q[1] = *(const uint4*)(p + 16);
    return f.v;
}

__device__ inline v8f wmma_f16(v16h a, v16h b, v8f c) {
    return __builtin_amdgcn_wmma_f32_16x16x32_f16(false, a, false, b, (short)0, c,
                                                  false, false);
}

// ============================================================
// Projection GEMM: OUT[o][n] = act(sum_c Aw[o][c]*X[c][n] + bias[o])
// X is channel-major f32 (row stride HW), KDIM = 256.
// MODE 0: f32 channel-major output (hidden).  MODE 1: f16 pixel-major output.
// Block tile 128(M) x 128(N), 8 waves (2x4), wave tile 64x32, WMMA 16x16x32.
// ============================================================
template<int MODE, int RELU>
__global__ __launch_bounds__(256) void proj_gemm_kernel(
    const float* __restrict__ Aw, const float* __restrict__ bias,
    const float* __restrict__ Xbase, int Mrows,
    float* __restrict__ outF32, _Float16* __restrict__ out16)
{
    __shared__ _Float16 As[128 * LDS_STRIDE];
    __shared__ _Float16 Bs[128 * LDS_STRIDE];

    const int tid  = threadIdx.x;
    const int lane = tid & 31;
    const int wave = tid >> 5;
    const int wm   = wave >> 2;        // 0..1
    const int wn   = wave & 3;         // 0..3
    const int nBlock = blockIdx.x * 128;
    const int mBlock = blockIdx.y * 128;
    const int img    = blockIdx.z;

    const float* X = Xbase + (size_t)img * CDIM * HW;

    FragC acc[4][2];
#pragma unroll
    for (int i = 0; i < 4; i++)
#pragma unroll
        for (int j = 0; j < 2; j++)
#pragma unroll
            for (int r = 0; r < 8; r++) acc[i][j].f[r] = 0.0f;

    for (int kb = 0; kb < CDIM; kb += 64) {
        // ---- A tile: 128 rows x 64 k (f32 -> f16), float4 granularity ----
#pragma unroll
        for (int it = 0; it < 8; it++) {
            int e  = tid + it * 256;          // 2048 float4s
            int r  = e >> 4;
            int c4 = (e & 15) << 2;
            int gm = mBlock + r;
            float4 v = make_float4(0.f, 0.f, 0.f, 0.f);
            if (gm < Mrows) v = *(const float4*)&Aw[(size_t)gm * CDIM + kb + c4];
            _Float16* d = &As[r * LDS_STRIDE + c4];
            d[0] = (_Float16)v.x; d[1] = (_Float16)v.y;
            d[2] = (_Float16)v.z; d[3] = (_Float16)v.w;
        }
        // ---- B tile: X[kb..kb+63][nBlock..+127] -> Bs[n][k] (transposed) ----
#pragma unroll
        for (int it = 0; it < 8; it++) {
            int e  = tid + it * 256;          // 2048 float4s (k-major over n)
            int k  = e >> 5;
            int n4 = (e & 31) << 2;
            float4 v = *(const float4*)&X[(size_t)(kb + k) * HW + nBlock + n4];
            Bs[(n4 + 0) * LDS_STRIDE + k] = (_Float16)v.x;
            Bs[(n4 + 1) * LDS_STRIDE + k] = (_Float16)v.y;
            Bs[(n4 + 2) * LDS_STRIDE + k] = (_Float16)v.z;
            Bs[(n4 + 3) * LDS_STRIDE + k] = (_Float16)v.w;
        }
        __syncthreads();

#pragma unroll
        for (int ks = 0; ks < 64; ks += 32) {
            v16h af[4], bf[2];
#pragma unroll
            for (int i = 0; i < 4; i++) af[i] = lds_frag(As, wm * 64 + i * 16, ks, lane);
#pragma unroll
            for (int j = 0; j < 2; j++) bf[j] = lds_frag(Bs, wn * 32 + j * 16, ks, lane);
#pragma unroll
            for (int i = 0; i < 4; i++)
#pragma unroll
                for (int j = 0; j < 2; j++)
                    acc[i][j].v = wmma_f16(af[i], bf[j], acc[i][j].v);
        }
        __syncthreads();
    }

    // ---- epilogue ----
    if (MODE == 0) {
        float* o32 = outF32 + (size_t)img * CDIM * HW;
#pragma unroll
        for (int i = 0; i < 4; i++) {
            int oBase = mBlock + wm * 64 + i * 16 + ((lane >> 4) << 3);
#pragma unroll
            for (int j = 0; j < 2; j++) {
                int n = nBlock + wn * 32 + j * 16 + (lane & 15);
#pragma unroll
                for (int r = 0; r < 8; r++) {
                    int o = oBase + r;
                    if (o < Mrows) {
                        float v = acc[i][j].f[r] + bias[o];
                        if (RELU) v = fmaxf(v, 0.0f);
                        o32[(size_t)o * HW + n] = v;
                    }
                }
            }
        }
    } else {
        _Float16* o16 = out16 + (size_t)img * HW * KD;
#pragma unroll
        for (int i = 0; i < 4; i++) {
            int oBase = mBlock + wm * 64 + i * 16 + ((lane >> 4) << 3);
            if (oBase + 8 <= Mrows) {
#pragma unroll
                for (int j = 0; j < 2; j++) {
                    int n = nBlock + wn * 32 + j * 16 + (lane & 15);
                    Pack8h pk;
#pragma unroll
                    for (int r = 0; r < 8; r++) {
                        float v = acc[i][j].f[r] + bias[oBase + r];
                        if (RELU) v = fmaxf(v, 0.0f);
                        pk.h[r] = (_Float16)v;
                    }
                    *(uint4*)&o16[(size_t)n * KD + oBase] = pk.u;
                }
            }
        }
    }
}

// ============================================================
// Correlation GEMM: S = Q(8448x64) * K(8448x64)^T  (f16 WMMA),
// fused epilogue  raw[n*HW + m] = -sqrt(max(fa2[m]+fb2[n]-2S, 0))
// (note transposed store: channel = key pixel n, spatial = query pixel m)
// Tiles are pure byte copies -> CDNA5 async global->LDS path.
// ============================================================
__global__ __launch_bounds__(256) void corr_gemm_kernel(
    const _Float16* __restrict__ q16, const _Float16* __restrict__ k16,
    const float* __restrict__ fa2, const float* __restrict__ fb2,
    float* __restrict__ raw)
{
    __shared__ _Float16 As[128 * LDS_STRIDE];
    __shared__ _Float16 Bs[128 * LDS_STRIDE];

    const int tid  = threadIdx.x;
    const int lane = tid & 31;
    const int wave = tid >> 5;
    const int wm   = wave >> 2;
    const int wn   = wave & 3;
    const int nBlock = blockIdx.x * 128;   // key pixel / output channel
    const int mBlock = blockIdx.y * 128;   // query pixel / output spatial

    // direct 16B copies: rows of 64 halves (pixel-major layout == WMMA source)
#if USE_ASYNC_LDS
#pragma unroll
    for (int it = 0; it < 4; it++) {
        int e  = tid + it * 256;           // 1024 uint4s per tile
        int r  = e >> 3;
        int c8 = (e & 7) << 3;
        async_copy16(&q16[(size_t)(mBlock + r) * KD + c8], &As[r * LDS_STRIDE + c8]);
        async_copy16(&k16[(size_t)(nBlock + r) * KD + c8], &Bs[r * LDS_STRIDE + c8]);
    }
    async_wait0();
    __syncthreads();
#else
#pragma unroll
    for (int it = 0; it < 4; it++) {
        int e  = tid + it * 256;
        int r  = e >> 3;
        int c8 = (e & 7) << 3;
        *(uint4*)&As[r * LDS_STRIDE + c8] =
            *(const uint4*)&q16[(size_t)(mBlock + r) * KD + c8];
        *(uint4*)&Bs[r * LDS_STRIDE + c8] =
            *(const uint4*)&k16[(size_t)(nBlock + r) * KD + c8];
    }
    __syncthreads();
#endif

    FragC acc[4][2];
#pragma unroll
    for (int i = 0; i < 4; i++)
#pragma unroll
        for (int j = 0; j < 2; j++)
#pragma unroll
            for (int r = 0; r < 8; r++) acc[i][j].f[r] = 0.0f;

#pragma unroll
    for (int ks = 0; ks < 64; ks += 32) {
        v16h af[4], bf[2];
#pragma unroll
        for (int i = 0; i < 4; i++) af[i] = lds_frag(As, wm * 64 + i * 16, ks, lane);
#pragma unroll
        for (int j = 0; j < 2; j++) bf[j] = lds_frag(Bs, wn * 32 + j * 16, ks, lane);
#pragma unroll
        for (int i = 0; i < 4; i++)
#pragma unroll
            for (int j = 0; j < 2; j++)
                acc[i][j].v = wmma_f16(af[i], bf[j], acc[i][j].v);
    }

#pragma unroll
    for (int i = 0; i < 4; i++) {
        int mBaseL = mBlock + wm * 64 + i * 16 + ((lane >> 4) << 3);
        float4 a2lo = *(const float4*)&fa2[mBaseL];
        float4 a2hi = *(const float4*)&fa2[mBaseL + 4];
        float a2[8] = {a2lo.x, a2lo.y, a2lo.z, a2lo.w, a2hi.x, a2hi.y, a2hi.z, a2hi.w};
#pragma unroll
        for (int j = 0; j < 2; j++) {
            int n = nBlock + wn * 32 + j * 16 + (lane & 15);
            float b2 = fb2[n];
            float ov[8];
#pragma unroll
            for (int r = 0; r < 8; r++) {
                float d2 = a2[r] + b2 - 2.0f * acc[i][j].f[r];
                ov[r] = -sqrtf(fmaxf(d2, 0.0f));
            }
            float* dst = raw + (size_t)n * HW + mBaseL;
            *(float4*)(dst)     = make_float4(ov[0], ov[1], ov[2], ov[3]);
            *(float4*)(dst + 4) = make_float4(ov[4], ov[5], ov[6], ov[7]);
        }
    }
}

// ---- sum of squares of f16 64-vectors (fa2 / fb2) ----
__global__ __launch_bounds__(256) void sumsq_kernel(const _Float16* __restrict__ v16,
                                                    float* __restrict__ out, int total)
{
    int g = blockIdx.x * 256 + threadIdx.x;
    if (g >= total) return;
    const _Float16* p = v16 + (size_t)g * KD;
    float s = 0.0f;
#pragma unroll
    for (int k = 0; k < KD; k += 8) {
        Pack8h pk;
        pk.u = *(const uint4*)&p[k];
#pragma unroll
        for (int t = 0; t < 8; t++) { float x = (float)pk.h[t]; s += x * x; }
    }
    out[g] = s;
}

// ---- last correlation channel: -|q| ----
__global__ __launch_bounds__(256) void last_chan_kernel(const float* __restrict__ fa2s,
                                                        float* __restrict__ raw)
{
    int a = blockIdx.x * 256 + threadIdx.x;
    if (a < HW) raw[(size_t)(KCH - 1) * HW + a] = -sqrtf(fa2s[a]);
}

// ---- depthwise 3x3 smooth (SAME, cross-correlation) ----
// Output is written once, > L2 capacity -> non-temporal store keeps the
// raw-volume rows (which ARE re-read 9x) resident instead.
__global__ __launch_bounds__(256) void smooth_kernel(const float* __restrict__ raw,
                                                     const float* __restrict__ smw,
                                                     const float* __restrict__ smb,
                                                     float* __restrict__ out)
{
    int g = blockIdx.x * 256 + threadIdx.x;          // < KCH*HW (fits int32)
    if (g >= KCH * HW) return;
    int ch = g / HW;
    int a  = g - ch * HW;
    int y  = a / WW;
    int x  = a - y * WW;
    const float* wv = smw + ch * 9;
    const float* pc = raw + (size_t)ch * HW;
    float accv = smb[ch];
#pragma unroll
    for (int dy = -1; dy <= 1; dy++) {
        int yy = y + dy;
        if (yy < 0 || yy >= HH) continue;
#pragma unroll
        for (int dx = -1; dx <= 1; dx++) {
            int xx = x + dx;
            if (xx < 0 || xx >= WW) continue;
            accv += wv[(dy + 1) * 3 + (dx + 1)] * pc[yy * WW + xx];
        }
    }
    __builtin_nontemporal_store(accv, &out[g]);
}

// ---- argmax over channels per spatial pixel (coalesced channel scan) ----
__global__ __launch_bounds__(256) void argmax_kernel(const float* __restrict__ vol,
                                                     int* __restrict__ idx,
                                                     float* __restrict__ mxv)
{
    int a = blockIdx.x * 256 + threadIdx.x;
    if (a >= HW) return;
    float best = -INFINITY;
    int bi = 0;
    for (int ch = 0; ch < KCH; ch++) {
        float v = __builtin_nontemporal_load(&vol[(size_t)ch * HW + a]);
        if (v > best) { best = v; bi = ch; }
    }
    idx[a] = bi;
    mxv[a] = best;
}

// ---- fuse (grid_sample + weight*mask + max) and cls/reg heads ----
__global__ __launch_bounds__(256) void fuse_head_kernel(
    const float* __restrict__ feats,
    const float* __restrict__ cls_w, const float* __restrict__ cls_b,
    const float* __restrict__ reg_w, const float* __restrict__ reg_b,
    const int* __restrict__ idxArr, const float* __restrict__ mxvArr,
    float* __restrict__ psm, float* __restrict__ rm)
{
    __shared__ float fuse[CDIM];
    const int a = blockIdx.x;        // spatial pixel
    const int p = blockIdx.y;        // pair
    const int c = threadIdx.x;       // channel
    const int s = 2 * p, i = 2 * p + 1;

    int   id   = idxArr[p * HW + a];
    float wmax = mxvArr[p * HW + a];
    bool  mask = (id == KCH - 1);
    int x = mask ? 0 : (id % WW);
    int y = mask ? 0 : (id / WW);

    float xn = 2.0f * ((float)x / (float)WW - 0.5f);
    float yn = 2.0f * ((float)y / (float)HH - 0.5f);
    float gx = (xn + 1.0f) * (WW * 0.5f) - 0.5f;
    float gy = (yn + 1.0f) * (HH * 0.5f) - 0.5f;
    float x0 = floorf(gx), y0 = floorf(gy);
    float wx = gx - x0,    wy = gy - y0;
    int x0i = (int)x0, y0i = (int)y0;

    const float* img = feats + ((size_t)i * CDIM + c) * HW;
    auto tap = [&](int ix, int iy, float wgt) -> float {
        bool valid = (ix >= 0) && (ix < WW) && (iy >= 0) && (iy < HH);
        int cx = ix < 0 ? 0 : (ix > WW - 1 ? WW - 1 : ix);
        int cy = iy < 0 ? 0 : (iy > HH - 1 ? HH - 1 : iy);
        return img[cy * WW + cx] * (valid ? wgt : 0.0f);
    };
    float gs = tap(x0i,     y0i,     (1 - wx) * (1 - wy))
             + tap(x0i + 1, y0i,     wx * (1 - wy))
             + tap(x0i,     y0i + 1, (1 - wx) * wy)
             + tap(x0i + 1, y0i + 1, wx * wy);
    float wf = gs * wmax * (mask ? 1.0f : 0.0f);
    float fv = fmaxf(feats[((size_t)s * CDIM + c) * HW + a], wf);
    fuse[c] = fv;
    __syncthreads();

    if (c < ANC + NREG) {
        const float* wrow;
        float b;
        float* outp;
        if (c < ANC) {
            wrow = cls_w + c * CDIM; b = cls_b[c];
            outp = psm + ((size_t)p * ANC + c) * HW + a;
        } else {
            int o = c - ANC;
            wrow = reg_w + o * CDIM; b = reg_b[o];
            outp = rm + ((size_t)p * NREG + o) * HW + a;
        }
        float sacc = b;
        for (int c2 = 0; c2 < CDIM; c2++) sacc += wrow[c2] * fuse[c2];
        *outp = sacc;
    }
}

// ============================================================
extern "C" void kernel_launch(void* const* d_in, const int* in_sizes, int n_in,
                              void* d_out, int out_size, void* d_ws, size_t ws_size,
                              hipStream_t stream) {
    (void)in_sizes; (void)n_in; (void)out_size; (void)ws_size;
    const float* feats = (const float*)d_in[0];
    const float* q_w   = (const float*)d_in[1];
    const float* q_b   = (const float*)d_in[2];
    const float* qk_w  = (const float*)d_in[3];
    const float* qk_b  = (const float*)d_in[4];
    const float* k_w   = (const float*)d_in[5];
    const float* k_b   = (const float*)d_in[6];
    const float* sm_w  = (const float*)d_in[7];
    const float* sm_b  = (const float*)d_in[8];
    const float* cls_w = (const float*)d_in[9];
    const float* cls_b = (const float*)d_in[10];
    const float* reg_w = (const float*)d_in[11];
    const float* reg_b = (const float*)d_in[12];
    // record_len (d_in[13]) is fixed [2,2] -> pairs (0,1) and (2,3)
    float* out = (float*)d_out;

    // ---- workspace layout (hidden aliases raw: disjoint lifetimes) ----
    char* ws = (char*)d_ws;
    float*    raw    = (float*)ws;                 // KCH*HW f32  (285.5 MB)
    float*    hidden = (float*)ws;                 // 4*256*HW f32 (34.6 MB)
    size_t off = (size_t)KCH * HW * sizeof(float);
    _Float16* q16 = (_Float16*)(ws + off); off += (size_t)4 * HW * KD * 2;
    _Float16* k16 = (_Float16*)(ws + off); off += (size_t)4 * HW * KD * 2;
    float* fa2 = (float*)(ws + off); off += (size_t)4 * HW * 4;
    float* fb2 = (float*)(ws + off); off += (size_t)4 * HW * 4;
    int*   idx = (int*)(ws + off);   off += (size_t)2 * HW * 4;
    float* mxv = (float*)(ws + off);

    const dim3 blk(256);

    // 1) hidden = relu(q_w @ X + q_b)         (256 -> 256, f32 out)
    proj_gemm_kernel<0, 1><<<dim3(66, 2, 4), blk, 0, stream>>>(
        q_w, q_b, feats, CDIM, hidden, nullptr);
    // 2) queries = qk_w @ hidden + qk_b       (256 -> 64, f16 pixel-major)
    proj_gemm_kernel<1, 0><<<dim3(66, 1, 4), blk, 0, stream>>>(
        qk_w, qk_b, hidden, KD, nullptr, q16);
    // 3) keys = relu(k_w @ X + k_b)           (256 -> 64, f16 pixel-major)
    proj_gemm_kernel<1, 1><<<dim3(66, 1, 4), blk, 0, stream>>>(
        k_w, k_b, feats, KD, nullptr, k16);
    // 4) squared norms
    sumsq_kernel<<<dim3(4 * HW / 256), blk, 0, stream>>>(q16, fa2, 4 * HW);
    sumsq_kernel<<<dim3(4 * HW / 256), blk, 0, stream>>>(k16, fb2, 4 * HW);

    const size_t RM_OFF   = (size_t)2 * ANC * HW;
    const size_t CORR_OFF = RM_OFF + (size_t)2 * NREG * HW;

    for (int p = 0; p < 2; p++) {
        int s = 2 * p, i = 2 * p + 1;
        corr_gemm_kernel<<<dim3(66, 66), blk, 0, stream>>>(
            q16 + (size_t)s * HW * KD, k16 + (size_t)i * HW * KD,
            fa2 + (size_t)s * HW, fb2 + (size_t)i * HW, raw);
        last_chan_kernel<<<dim3(HW / 256), blk, 0, stream>>>(fa2 + (size_t)s * HW, raw);
        smooth_kernel<<<dim3((KCH * HW + 255) / 256), blk, 0, stream>>>(
            raw, sm_w, sm_b, out + CORR_OFF + (size_t)p * KCH * HW);
        argmax_kernel<<<dim3(HW / 256), blk, 0, stream>>>(
            out + CORR_OFF + (size_t)p * KCH * HW, idx + p * HW, mxv + p * HW);
    }

    fuse_head_kernel<<<dim3(HW, 2), blk, 0, stream>>>(
        feats, cls_w, cls_b, reg_w, reg_b, idx, mxv, out, out + RM_OFF);
}
